// NeRF_21526376088308
// MI455X (gfx1250) — matrix operand order repaired and checked by reading, not compiled, not run on hardware
//
#include <hip/hip_runtime.h>

// ---------------------------------------------------------------------------
// NeRF coarse->fine pipeline for MI455X (gfx1250, wave32, WMMA).
// Compute-bound (~90 GFLOP vs ~40MB traffic) => f16 WMMA 16x16x32, f32 accum.
// Round 3: 8 waves/block sharing one LDS weight copy (dynamic LDS, 94KB),
// per-wave private activation buffers, no intra-loop barriers -> ~6 waves/SIMD
// for latency hiding. global_prefetch_b8 on the fine-point stream.
// ---------------------------------------------------------------------------

typedef __attribute__((ext_vector_type(16))) _Float16 v16h;
typedef __attribute__((ext_vector_type(8)))  _Float16 v8h;
typedef __attribute__((ext_vector_type(8)))  float    v8f;

#define NEARP 0.1f
#define FARP  100.0f
#define NRAYS 8192
#define NCS   64     // coarse samples per ray
#define NFS   128    // fine samples per ray
#define SHLD  136    // activation stride (halfs): 272B = 17*16B, keeps runs aligned
#define WAVES 8      // waves per block sharing one weight copy

// dynamic LDS layout (halfs):
//   [0      , 4096 )  sW1f : L1 fragments, 8 n-tiles
//   [4096   , 20480)  sW2f : L2 fragments, 4 ks x 8 t
//   [20480  , 28672)  sW3f : L3 fragments, 4 ks x 4 t
//   [28672  , 29696)  sW4f : L4 fragments, 2 ks
//   [29696  , +WAVES*16*SHLD)  per-wave activation buffers
#define SMEM_W1F 0
#define SMEM_W2F (8 * 512)
#define SMEM_W3F (SMEM_W2F + 32 * 512)
#define SMEM_W4F (SMEM_W3F + 16 * 512)
#define SMEM_H   (SMEM_W4F + 2 * 512)
#define SMEM_HALFS (SMEM_H + WAVES * 16 * SHLD)

__device__ __forceinline__ float sigmoid_f(float x) {
  return __builtin_amdgcn_rcpf(1.0f + __expf(-x));   // v_exp + v_rcp
}
__device__ __forceinline__ float silu_f(float x) { return x * sigmoid_f(x); }

__device__ __forceinline__ v8f wmma16(v16h a, v16h b, v8f c) {
  // v_wmma_f32_16x16x32_f16: D = A(16x32 f16) * B(32x16 f16) + C(16x16 f32)
  return __builtin_amdgcn_wmma_f32_16x16x32_f16(false, a, false, b, (short)0, c,
                                                false, false);
}

// B-fragment from pre-swizzled LDS: 32 contiguous bytes per lane -> 2x ds_load_b128.
__device__ __forceinline__ v16h load_frag16(const _Float16* base, int frag, int lane) {
  return *(const v16h*)(base + ((frag * 32 + lane) << 4));
}

// A-fragment (16x32 f16) from row-major LDS H[m][k] (stride SHLD).
// Layout used here (contiguous K, same as B staging): lane m=l&15, kh=l>>4; halfs 0-15 = K 16*kh+0..15.
// Both runs are 8 contiguous halfs, 16B aligned -> 2x ds_load_b128.
__device__ __forceinline__ v16h load_a_frag(const _Float16* H, int k0, int lane) {
  const int m  = lane & 15;
  const int kh = lane >> 4;
  v8h lo = *(const v8h*)(H + m * SHLD + k0 + 16 * kh);
  v8h hi = *(const v8h*)(H + m * SHLD + k0 + 16 * kh + 8);
  return __builtin_shufflevector(lo, hi, 0, 1, 2, 3, 4, 5, 6, 7,
                                 8, 9, 10, 11, 12, 13, 14, 15);
}

// Positional-encoding feature k of point (x,y,z); k in [0,32), features >=27 pad to 0.
__device__ __forceinline__ float pe_feat(int k, float x, float y, float z) {
  if (k >= 27) return 0.0f;
  if (k < 3) return (k == 0) ? x : ((k == 1) ? y : z);
  int j = k - 3;
  int i = j / 6;
  int r = j % 6;
  int c = r % 3;
  float pc = (c == 0) ? x : ((c == 1) ? y : z);
  float f = 3.14159265358979f * (float)(1 << i);
  return (r < 3) ? __sinf(f * pc) : __cosf(f * pc);
}

// ---------------------------------------------------------------------------
// Fused MLP kernel: 8 waves/block, one shared LDS weight copy (f16, pre-
// swizzled into WMMA B-fragment order), per-wave activation scratch.
// mode 0 (coarse): points from rays/z_coarse; writes sigma only.
// mode 1 (fine)  : points from pts_in; writes sigma + sigmoid(rgb).
// Each wave processes independent 16-point tiles; 58 v_wmma per tile.
// ---------------------------------------------------------------------------
__global__ __launch_bounds__(32 * WAVES) void nerf_mlp_kernel(
    const float* __restrict__ W1, const float* __restrict__ b1,
    const float* __restrict__ W2, const float* __restrict__ b2,
    const float* __restrict__ W3, const float* __restrict__ b3,
    const float* __restrict__ W4, const float* __restrict__ b4,
    const float* __restrict__ ro, const float* __restrict__ rd,
    const float* __restrict__ u_strat,
    const float* __restrict__ pts_in,
    float* __restrict__ sigma_out,
    float* __restrict__ rgb_out,
    int mode, int ntiles)
{
  extern __shared__ _Float16 smem[];
  _Float16* sW1f = smem + SMEM_W1F;
  _Float16* sW2f = smem + SMEM_W2F;
  _Float16* sW3f = smem + SMEM_W3F;
  _Float16* sW4f = smem + SMEM_W4F;

  const int tid  = threadIdx.x;
  const int lane = tid & 31;
  const int wave = tid >> 5;
  _Float16* sH = smem + SMEM_H + wave * (16 * SHLD);

  // ---- stage weights (f32 -> f16) swizzled into B-fragment order ----
  // element (frag f, lane l, j): K = k0(f) + 16*(l>>4) + j, N = n0(f) + (l&15)
  for (int i = tid; i < 8 * 512; i += 32 * WAVES) {
    int f = i >> 9, l = (i >> 4) & 31, j = i & 15;
    int k = ((l >> 4) << 4) + j;                 // 0..31 (27..31 zero-pad)
    int n = f * 16 + (l & 15);
    sW1f[i] = (_Float16)((k < 27) ? W1[k * 128 + n] : 0.0f);
  }
  for (int i = tid; i < 32 * 512; i += 32 * WAVES) {
    int f = i >> 9, l = (i >> 4) & 31, j = i & 15;
    int k = (f >> 3) * 32 + ((l >> 4) << 4) + j;
    int n = (f & 7) * 16 + (l & 15);
    sW2f[i] = (_Float16)W2[k * 128 + n];
  }
  for (int i = tid; i < 16 * 512; i += 32 * WAVES) {
    int f = i >> 9, l = (i >> 4) & 31, j = i & 15;
    int k = (f >> 2) * 32 + ((l >> 4) << 4) + j;
    int n = (f & 3) * 16 + (l & 15);
    sW3f[i] = (_Float16)W3[k * 64 + n];
  }
  for (int i = tid; i < 2 * 512; i += 32 * WAVES) {
    int f = i >> 9, l = (i >> 4) & 31, j = i & 15;
    int k = f * 32 + ((l >> 4) << 4) + j;
    int n = l & 15;                               // 4..15 zero-pad
    sW4f[i] = (_Float16)((n < 4) ? W4[k * 4 + n] : 0.0f);
  }
  __syncthreads();   // weights visible to all waves; no further barriers needed

  const int mrow = lane & 15;   // A-frag row / point-in-tile
  const int kh   = lane >> 4;   // A-frag K-half / C-frag M-group
  const int ncol = lane & 15;   // C-frag column

  // per-lane biases (fixed across tiles)
  float bias1[8], bias2[8], bias3[4];
#pragma unroll
  for (int t = 0; t < 8; ++t) bias1[t] = b1[16 * t + ncol];
#pragma unroll
  for (int t = 0; t < 8; ++t) bias2[t] = b2[16 * t + ncol];
#pragma unroll
  for (int t = 0; t < 4; ++t) bias3[t] = b3[16 * t + ncol];
  const float bias4 = (ncol < 4) ? b4[ncol] : 0.0f;

  const v8f zero8 = {0.f, 0.f, 0.f, 0.f, 0.f, 0.f, 0.f, 0.f};

  const int slot   = blockIdx.x * WAVES + wave;   // this wave's tile stream
  const int nslots = gridDim.x * WAVES;

  for (int tile = slot; tile < ntiles; tile += nslots) {
    // ---- point coordinates for this lane's A-frag row ----
    const int p = tile * 16 + mrow;
    float px, py, pz;
    if (mode == 0) {
      int r = p >> 6;            // ray
      int c = p & (NCS - 1);     // coarse sample
      float z = NEARP + ((float)c + u_strat[c]) * ((FARP - NEARP) / (float)NCS);
      px = ro[r * 3 + 0] + rd[r * 3 + 0] * z;
      py = ro[r * 3 + 1] + rd[r * 3 + 1] * z;
      pz = ro[r * 3 + 2] + rd[r * 3 + 2] * z;
    } else {
      px = pts_in[p * 3 + 0];
      py = pts_in[p * 3 + 1];
      pz = pts_in[p * 3 + 2];
      // prefetch next tile's points (global_prefetch_b8)
      int pn = (tile + nslots) * 16 + mrow;
      if (tile + nslots < ntiles) __builtin_prefetch(pts_in + pn * 3, 0, 1);
    }

    // ---- positional encoding straight into A-fragment registers ----
    v16h xa;
#pragma unroll
    for (int j = 0; j < 16; ++j) {
      int k = 16 * kh + j;
      xa[j] = (_Float16)pe_feat(k, px, py, pz);
    }

    // ---- layer 1: [16x32] @ [32x128] -> sH (silu) ----
#pragma unroll
    for (int t = 0; t < 8; ++t) {
      v16h bf = load_frag16(sW1f, t, lane);
      v8f acc = wmma16(xa, bf, zero8);
#pragma unroll
      for (int v = 0; v < 8; ++v)
        sH[(v + 8 * kh) * SHLD + 16 * t + ncol] = (_Float16)silu_f(acc[v] + bias1[t]);
    }

    // ---- layer 2: [16x128] @ [128x128] -> sH (silu) ----
    v8f acc2[8];
#pragma unroll
    for (int t = 0; t < 8; ++t) acc2[t] = zero8;
#pragma unroll
    for (int ks = 0; ks < 4; ++ks) {
      v16h a = load_a_frag(sH, 32 * ks, lane);
#pragma unroll
      for (int t = 0; t < 8; ++t) {
        v16h bf = load_frag16(sW2f, ks * 8 + t, lane);
        acc2[t] = wmma16(a, bf, acc2[t]);
      }
    }
#pragma unroll
    for (int t = 0; t < 8; ++t)
#pragma unroll
      for (int v = 0; v < 8; ++v)
        sH[(v + 8 * kh) * SHLD + 16 * t + ncol] = (_Float16)silu_f(acc2[t][v] + bias2[t]);

    // ---- layer 3: [16x128] @ [128x64] -> sH (silu) ----
    v8f acc3[4];
#pragma unroll
    for (int t = 0; t < 4; ++t) acc3[t] = zero8;
#pragma unroll
    for (int ks = 0; ks < 4; ++ks) {
      v16h a = load_a_frag(sH, 32 * ks, lane);
#pragma unroll
      for (int t = 0; t < 4; ++t) {
        v16h bf = load_frag16(sW3f, ks * 4 + t, lane);
        acc3[t] = wmma16(a, bf, acc3[t]);
      }
    }
#pragma unroll
    for (int t = 0; t < 4; ++t)
#pragma unroll
      for (int v = 0; v < 8; ++v)
        sH[(v + 8 * kh) * SHLD + 16 * t + ncol] = (_Float16)silu_f(acc3[t][v] + bias3[t]);

    // ---- layer 4: [16x64] @ [64x16(pad of 4)] ----
    v8f acc4 = zero8;
#pragma unroll
    for (int ks = 0; ks < 2; ++ks) {
      v16h a  = load_a_frag(sH, 32 * ks, lane);
      v16h bf = load_frag16(sW4f, ks, lane);
      acc4 = wmma16(a, bf, acc4);
    }

    // ---- emit outputs: ncol 0..2 -> rgb (fine only), ncol==3 -> sigma ----
#pragma unroll
    for (int v = 0; v < 8; ++v) {
      int pp = tile * 16 + v + 8 * kh;
      float o = acc4[v] + bias4;
      if (ncol == 3) {
        sigma_out[pp] = fmaxf(o, 0.0f);
      } else if (ncol < 3 && mode == 1) {
        rgb_out[pp * 3 + ncol] = sigmoid_f(o);
      }
    }
  }
}

// ---------------------------------------------------------------------------
// Kernel 2: per-ray inverse-CDF importance sampling.
// One thread per ray; per-thread CDF (63 bins) lives in LDS.
// ---------------------------------------------------------------------------
__global__ __launch_bounds__(128) void nerf_sample_kernel(
    const float* __restrict__ ro, const float* __restrict__ rd,
    const float* __restrict__ u_strat, const float* __restrict__ u_fine,
    const float* __restrict__ sigma_c,
    float* __restrict__ z_fine, float* __restrict__ pts_fine)
{
  __shared__ float scdf[128][NCS - 1];  // 128 * 63 * 4 = 31.5 KB
  const int r = blockIdx.x * 128 + threadIdx.x;
  float* cdf = scdf[threadIdx.x];

  const float step = (FARP - NEARP) / (float)NCS;
  float T = 1.0f, cum = 0.0f;
  float z_prev = NEARP + (0.0f + u_strat[0]) * step;
  float sg     = sigma_c[r * NCS + 0];
#pragma unroll 1
  for (int i = 0; i < NCS - 1; ++i) {
    float z_next = NEARP + ((float)(i + 1) + u_strat[i + 1]) * step;
    float dz    = z_next - z_prev;
    float alpha = 1.0f - __expf(-sg * dz);
    float w     = alpha * T + 1e-10f;
    T *= (1.0f - alpha + 1e-10f);
    cum += w;
    cdf[i] = cum;
    z_prev = z_next;
    sg = sigma_c[r * NCS + i + 1];
  }
  const float inv = 1.0f / cum;

  const float ox = ro[r * 3 + 0], oy = ro[r * 3 + 1], oz = ro[r * 3 + 2];
  const float dx = rd[r * 3 + 0], dy = rd[r * 3 + 1], dz3 = rd[r * 3 + 2];

#pragma unroll 1
  for (int s = 0; s < NFS; ++s) {
    float u = u_fine[r * NFS + s];
    // searchsorted (side='left') over normalized cdf
    int lo = 0, hi = NCS - 1;
    while (lo < hi) {
      int mid = (lo + hi) >> 1;
      if (cdf[mid] * inv < u) lo = mid + 1; else hi = mid;
    }
    int idx = (lo > NCS - 2) ? (NCS - 2) : lo;
    float z = (float)idx * ((FARP - NEARP) / (float)(NCS - 1)) + NEARP;
    z_fine[r * NFS + s] = z;
    int p = r * NFS + s;
    pts_fine[p * 3 + 0] = ox + dx * z;
    pts_fine[p * 3 + 1] = oy + dy * z;
    pts_fine[p * 3 + 2] = oz + dz3 * z;
  }
}

// ---------------------------------------------------------------------------
// Kernel 4: per-ray alpha compositing over fine samples.
// ---------------------------------------------------------------------------
__global__ __launch_bounds__(128) void nerf_render_kernel(
    const float* __restrict__ sigma_f, const float* __restrict__ rgb_f,
    const float* __restrict__ z_fine, float* __restrict__ rgb_out)
{
  const int r = blockIdx.x * 128 + threadIdx.x;
  float T = 1.0f, a0 = 0.0f, a1 = 0.0f, a2 = 0.0f;
#pragma unroll 1
  for (int s = 0; s < NFS - 1; ++s) {
    int p = r * NFS + s;
    float dz    = z_fine[p + 1] - z_fine[p];
    float alpha = 1.0f - __expf(-sigma_f[p] * dz);
    float w     = alpha * T;
    a0 += w * rgb_f[p * 3 + 0];
    a1 += w * rgb_f[p * 3 + 1];
    a2 += w * rgb_f[p * 3 + 2];
    T *= (1.0f - alpha + 1e-10f);
  }
  rgb_out[r * 3 + 0] = a0;
  rgb_out[r * 3 + 1] = a1;
  rgb_out[r * 3 + 2] = a2;
}

// ---------------------------------------------------------------------------
extern "C" void kernel_launch(void* const* d_in, const int* in_sizes, int n_in,
                              void* d_out, int out_size, void* d_ws, size_t ws_size,
                              hipStream_t stream) {
  const float* ro      = (const float*)d_in[0];   // [8192,3]
  const float* rd      = (const float*)d_in[1];   // [8192,3]
  // d_in[2] = viewdirs (unused by reference)
  const float* u_strat = (const float*)d_in[3];   // [64]
  const float* u_fine  = (const float*)d_in[4];   // [8192,128]
  const float* W1 = (const float*)d_in[5];  const float* b1 = (const float*)d_in[6];
  const float* W2 = (const float*)d_in[7];  const float* b2 = (const float*)d_in[8];
  const float* W3 = (const float*)d_in[9];  const float* b3 = (const float*)d_in[10];
  const float* W4 = (const float*)d_in[11]; const float* b4 = (const float*)d_in[12];

  // workspace layout (floats)
  float* wsf      = (float*)d_ws;
  float* sigma_c  = wsf;                                   // 8192*64
  float* z_fine   = wsf + NRAYS * NCS;                     // 8192*128
  float* rgb_fine = wsf + NRAYS * NCS + NRAYS * NFS;       // 8192*128*3

  // output layout (floats, reference return order)
  float* out      = (float*)d_out;
  float* rgb_rend = out;                                   // 8192*3
  float* sigma_f  = out + NRAYS * 3;                       // 8192*128
  float* pts_fine = out + NRAYS * 3 + NRAYS * NFS;         // 8192*128*3

  const size_t smem_bytes = (size_t)SMEM_HALFS * sizeof(_Float16);  // ~94 KB
  (void)hipFuncSetAttribute((const void*)nerf_mlp_kernel,
                            hipFuncAttributeMaxDynamicSharedMemorySize,
                            (int)smem_bytes);

  // 1) coarse MLP: sigma over 8192*64 points (32768 tiles of 16)
  nerf_mlp_kernel<<<256, 32 * WAVES, smem_bytes, stream>>>(
      W1, b1, W2, b2, W3, b3, W4, b4, ro, rd, u_strat,
      nullptr, sigma_c, nullptr, /*mode=*/0, (NRAYS * NCS) / 16);

  // 2) importance sampling -> z_fine (ws), pts_fine (d_out)
  nerf_sample_kernel<<<NRAYS / 128, 128, 0, stream>>>(
      ro, rd, u_strat, u_fine, sigma_c, z_fine, pts_fine);

  // 3) fine MLP: rgb (ws) + sigma (d_out) over 8192*128 points (65536 tiles)
  nerf_mlp_kernel<<<256, 32 * WAVES, smem_bytes, stream>>>(
      W1, b1, W2, b2, W3, b3, W4, b4, ro, rd, u_strat,
      pts_fine, sigma_f, rgb_fine, /*mode=*/1, (NRAYS * NFS) / 16);

  // 4) volume rendering -> rgb_rendered
  nerf_render_kernel<<<NRAYS / 128, 128, 0, stream>>>(
      sigma_f, rgb_fine, z_fine, rgb_rend);
}
